// CausalMultiHeadAttention_7645041787406
// MI455X (gfx1250) — matrix operand order repaired and checked by reading, not compiled, not run on hardware
//
#include <hip/hip_runtime.h>
#include <hip/hip_bf16.h>

#define D_MODEL   1024
#define NUM_HEADS 16
#define DK        64
#define BATCH     4
#define SEQ       2048
#define M_ROWS    (BATCH * SEQ)   // 8192

typedef __bf16 bf16_t;
typedef __attribute__((ext_vector_type(8)))  float  v8f;
typedef __attribute__((ext_vector_type(16))) __bf16 v16bf;
typedef __attribute__((ext_vector_type(8)))  __bf16 v8bf;

static __device__ inline v8f wmma_bf16(v16bf a, v16bf b, v8f c) {
  // 8 args: (neg_a, A, neg_b, B, c_mod, C, reuse_a, reuse_b)
  return __builtin_amdgcn_wmma_f32_16x16x32_bf16(false, a, false, b, (short)0, c,
                                                 false, false);
}

static __device__ inline v16bf join16(v8bf lo, v8bf hi) {
  union U { v8bf p[2]; v16bf v; } u;
  u.p[0] = lo; u.p[1] = hi;
  return u.v;
}

static __device__ inline v8f zero8() {
  v8f z;
#pragma unroll
  for (int j = 0; j < 8; ++j) z[j] = 0.f;
  return z;
}

// xor-shuffle within groups of 16 lanes (wave32) via ds_swizzle
template <int XMASK>
static __device__ inline float swz(float v) {
  int i = __builtin_bit_cast(int, v);
  int o = __builtin_amdgcn_ds_swizzle(i, (XMASK << 10) | 0x1f);
  return __builtin_bit_cast(float, o);
}

// ---------------------------------------------------------------------------
// fp32 -> bf16 conversion (grid-stride)
// ---------------------------------------------------------------------------
__global__ void cvt_f32_bf16(const float* __restrict__ in, bf16_t* __restrict__ out,
                             int n) {
  int i = blockIdx.x * blockDim.x + threadIdx.x;
  int stride = gridDim.x * blockDim.x;
  for (; i < n; i += stride) out[i] = (bf16_t)in[i];
}

// ---------------------------------------------------------------------------
// NT GEMM:  C[M,N] = A[M,K] * B[N,K]^T   (A, B bf16, K-contiguous both)
// mode 0: store bf16 head-split layout [B, H, S, DK]   (for Q/K/V)
// mode 1: store f32 flat [M, N]                        (final output)
// block = 256 threads = 8 waves (2 M-waves x 4 N-waves), wave tile 64x64
// grid  = (N/256, M/128)
// ---------------------------------------------------------------------------
__global__ __launch_bounds__(256) void gemm_bf16_nt(const bf16_t* __restrict__ A,
                                                    const bf16_t* __restrict__ Bw,
                                                    void* __restrict__ Cout,
                                                    int mode) {
  const int tid = threadIdx.x;
  const int w  = tid >> 5;
  const int l  = tid & 31;
  const int lh = l >> 4;   // lane half (0/1)
  const int r  = l & 15;   // row/col within 16
  const int wave_m = w & 1;
  const int wave_n = w >> 1;
  const int row0 = blockIdx.y * 128 + wave_m * 64;
  const int col0 = blockIdx.x * 256 + wave_n * 64;

  v8f acc[4][4];
#pragma unroll
  for (int tm = 0; tm < 4; ++tm)
#pragma unroll
    for (int tn = 0; tn < 4; ++tn) acc[tm][tn] = zero8();

  for (int kk = 0; kk < D_MODEL; kk += 32) {
    v16bf afr[4], bfr[4];
#pragma unroll
    for (int tm = 0; tm < 4; ++tm) {
      // A fragment: lane half lh holds k=kk+16*lh .. +15, contiguous 32B
      const bf16_t* p = A + (size_t)(row0 + tm * 16 + r) * D_MODEL + kk + 16 * lh;
      afr[tm] = *(const v16bf*)p;
    }
#pragma unroll
    for (int tn = 0; tn < 4; ++tn) {
      // B fragment: lane half lh holds k=kk+16*lh .. +15, contiguous 32B
      const bf16_t* p = Bw + (size_t)(col0 + tn * 16 + r) * D_MODEL + kk + 16 * lh;
      bfr[tn] = *(const v16bf*)p;
    }
#pragma unroll
    for (int tm = 0; tm < 4; ++tm)
#pragma unroll
      for (int tn = 0; tn < 4; ++tn)
        acc[tm][tn] = wmma_bf16(afr[tm], bfr[tn], acc[tm][tn]);
  }

  if (mode == 0) {
    bf16_t* out = (bf16_t*)Cout;  // [B, H, S, DK]
#pragma unroll
    for (int tm = 0; tm < 4; ++tm)
#pragma unroll
      for (int tn = 0; tn < 4; ++tn)
#pragma unroll
        for (int i = 0; i < 8; ++i) {
          int mrow = row0 + tm * 16 + i + 8 * lh;   // global row = b*SEQ + s
          int n    = col0 + tn * 16 + r;            // global col = h*DK + dk
          int b = mrow >> 11, s = mrow & (SEQ - 1);
          int h = n >> 6,    dk = n & (DK - 1);
          size_t idx = (((size_t)(b * NUM_HEADS + h) * SEQ + s) << 6) + dk;
          out[idx] = (bf16_t)acc[tm][tn][i];
        }
  } else {
    float* out = (float*)Cout;
#pragma unroll
    for (int tm = 0; tm < 4; ++tm)
#pragma unroll
      for (int tn = 0; tn < 4; ++tn)
#pragma unroll
        for (int i = 0; i < 8; ++i) {
          int mrow = row0 + tm * 16 + i + 8 * lh;
          int n    = col0 + tn * 16 + r;
          out[(size_t)mrow * D_MODEL + n] = acc[tm][tn][i];
        }
  }
}

// ---------------------------------------------------------------------------
// Flash attention, causal. One block = one (b,h) and a 64-query tile.
// 4 waves x 16 query rows each. K tile and V^T tile staged in LDS (shared by
// waves); P bounced through per-wave LDS for C-layout -> A-layout conversion.
// grid = (B*H, SEQ/64), block = 128
// ---------------------------------------------------------------------------
__global__ __launch_bounds__(128) void attn_flash(const bf16_t* __restrict__ Q,
                                                  const bf16_t* __restrict__ K,
                                                  const bf16_t* __restrict__ V,
                                                  bf16_t* __restrict__ O) {
  __shared__ alignas(64) bf16_t ldsK[64 * 64];      // [kv][d]
  __shared__ alignas(64) bf16_t ldsV[64 * 64];      // transposed: [d][kv]
  __shared__ alignas(64) bf16_t ldsP[4][16 * 64];   // per-wave P tile

  const int tid = threadIdx.x;
  const int w  = tid >> 5;
  const int l  = tid & 31;
  const int lh = l >> 4;
  const int r  = l & 15;

  const int bh = blockIdx.x;            // b*H + h
  const int b  = bh >> 4;
  const int h  = bh & 15;
  const int qt = blockIdx.y;
  const int q_base = qt * 64 + w * 16;  // first query row of this wave

  const bf16_t* Qg = Q + (size_t)bh * SEQ * DK;
  const bf16_t* Kg = K + (size_t)bh * SEQ * DK;
  const bf16_t* Vg = V + (size_t)bh * SEQ * DK;

  // Q fragments (16 rows x 64 dk -> 2 k-steps), loaded once
  v16bf aQ[2];
#pragma unroll
  for (int ks = 0; ks < 2; ++ks) {
    const bf16_t* p = Qg + (size_t)(q_base + r) * DK + ks * 32 + 16 * lh;
    aQ[ks] = *(const v16bf*)p;
  }

  float m_run[8], l_run[8];
#pragma unroll
  for (int i = 0; i < 8; ++i) { m_run[i] = -1e30f; l_run[i] = 0.f; }
  v8f o_acc[4];
#pragma unroll
  for (int tn = 0; tn < 4; ++tn) o_acc[tn] = zero8();

  for (int kt = 0; kt <= qt; ++kt) {
    // ---- stage K tile straight, V tile transposed ----
    for (int c = tid; c < 512; c += 128) {
      int row = c >> 3, col8 = (c & 7) << 3;
      *(v8bf*)(&ldsK[row * 64 + col8]) =
          *(const v8bf*)(Kg + (size_t)(kt * 64 + row) * DK + col8);
    }
    for (int c = tid; c < 512; c += 128) {
      int kv = c >> 3, d8 = (c & 7) << 3;
      v8bf vv = *(const v8bf*)(Vg + (size_t)(kt * 64 + kv) * DK + d8);
#pragma unroll
      for (int j = 0; j < 8; ++j) ldsV[(d8 + j) * 64 + kv] = vv[j];
    }
    if (kt < qt) {  // hint next tiles into cache (global_prefetch_b8)
      __builtin_prefetch(Kg + (size_t)(kt + 1) * 64 * DK + tid * 32, 0, 1);
      __builtin_prefetch(Vg + (size_t)(kt + 1) * 64 * DK + tid * 32, 0, 1);
    }
    __syncthreads();

    // ---- scores = Q * K^T ----
    v8f sc[4];
#pragma unroll
    for (int tn = 0; tn < 4; ++tn) sc[tn] = zero8();
#pragma unroll
    for (int tn = 0; tn < 4; ++tn)
#pragma unroll
      for (int ks = 0; ks < 2; ++ks) {
        v16bf bK = *(const v16bf*)(&ldsK[(tn * 16 + r) * 64 + ks * 32 + 16 * lh]);
        sc[tn] = wmma_bf16(aQ[ks], bK, sc[tn]);
      }

    // ---- scale + causal mask (diagonal tile only) ----
    const bool diag = (kt == qt);
#pragma unroll
    for (int tn = 0; tn < 4; ++tn) {
      int n_glob = kt * 64 + tn * 16 + r;
#pragma unroll
      for (int i = 0; i < 8; ++i) {
        float v = sc[tn][i] * 0.125f;   // 1/sqrt(64)
        if (diag) {
          int m_glob = q_base + i + 8 * lh;
          if (n_glob > m_glob) v = -1e30f;
        }
        sc[tn][i] = v;
      }
    }

    // ---- online softmax (row reductions across 16 lanes) ----
#pragma unroll
    for (int i = 0; i < 8; ++i) {
      float v = fmaxf(fmaxf(sc[0][i], sc[1][i]), fmaxf(sc[2][i], sc[3][i]));
      v = fmaxf(v, swz<1>(v)); v = fmaxf(v, swz<2>(v));
      v = fmaxf(v, swz<4>(v)); v = fmaxf(v, swz<8>(v));
      float m_new = fmaxf(m_run[i], v);
      float s = 0.f;
#pragma unroll
      for (int tn = 0; tn < 4; ++tn) {
        float p = __expf(sc[tn][i] - m_new);
        sc[tn][i] = p;
        s += p;
      }
      s += swz<1>(s); s += swz<2>(s); s += swz<4>(s); s += swz<8>(s);
      float corr = __expf(m_run[i] - m_new);
      l_run[i] = l_run[i] * corr + s;
      m_run[i] = m_new;
#pragma unroll
      for (int tn = 0; tn < 4; ++tn) o_acc[tn][i] *= corr;
    }

    // ---- P (C-layout) -> LDS -> A-layout fragments ----
#pragma unroll
    for (int tn = 0; tn < 4; ++tn)
#pragma unroll
      for (int i = 0; i < 8; ++i)
        ldsP[w][(i + 8 * lh) * 64 + tn * 16 + r] = (bf16_t)sc[tn][i];

    // ---- O += P * V ----
#pragma unroll
    for (int ks = 0; ks < 2; ++ks) {
      const bf16_t* p = &ldsP[w][r * 64 + ks * 32 + 16 * lh];
      v16bf aP = *(const v16bf*)p;
#pragma unroll
      for (int tn = 0; tn < 4; ++tn) {
        v16bf bV = *(const v16bf*)(&ldsV[(tn * 16 + r) * 64 + ks * 32 + 16 * lh]);
        o_acc[tn] = wmma_bf16(aP, bV, o_acc[tn]);
      }
    }
    __syncthreads();
  }

  // ---- normalize, store O as [B, S, D_MODEL] bf16 ----
  float inv[8];
#pragma unroll
  for (int i = 0; i < 8; ++i) inv[i] = 1.f / l_run[i];
#pragma unroll
  for (int tn = 0; tn < 4; ++tn)
#pragma unroll
    for (int i = 0; i < 8; ++i) {
      int s_glob = q_base + i + 8 * lh;
      int col = h * DK + tn * 16 + r;
      O[(size_t)(b * SEQ + s_glob) * D_MODEL + col] = (bf16_t)(o_acc[tn][i] * inv[i]);
    }
}

// ---------------------------------------------------------------------------
extern "C" void kernel_launch(void* const* d_in, const int* in_sizes, int n_in,
                              void* d_out, int out_size, void* d_ws, size_t ws_size,
                              hipStream_t stream) {
  (void)in_sizes; (void)n_in; (void)out_size; (void)ws_size;

  const float* x  = (const float*)d_in[0];
  const float* wq = (const float*)d_in[1];
  const float* wk = (const float*)d_in[2];
  const float* wv = (const float*)d_in[3];
  const float* wo = (const float*)d_in[4];

  const size_t XE = (size_t)M_ROWS * D_MODEL;   // 8388608 elements
  const size_t WE = (size_t)D_MODEL * D_MODEL;  // 1048576 elements

  char* p = (char*)d_ws;
  bf16_t* xb  = (bf16_t*)p; p += XE * 2;
  bf16_t* wqb = (bf16_t*)p; p += WE * 2;
  bf16_t* wkb = (bf16_t*)p; p += WE * 2;
  bf16_t* wvb = (bf16_t*)p; p += WE * 2;
  bf16_t* wob = (bf16_t*)p; p += WE * 2;
  bf16_t* Qb  = (bf16_t*)p; p += XE * 2;
  bf16_t* Kb  = (bf16_t*)p; p += XE * 2;
  bf16_t* Vb  = (bf16_t*)p; p += XE * 2;
  bf16_t* Ob  = (bf16_t*)p; p += XE * 2;        // total ~92.3 MB

  cvt_f32_bf16<<<2048, 256, 0, stream>>>(x,  xb,  (int)XE);
  cvt_f32_bf16<<<512,  256, 0, stream>>>(wq, wqb, (int)WE);
  cvt_f32_bf16<<<512,  256, 0, stream>>>(wk, wkb, (int)WE);
  cvt_f32_bf16<<<512,  256, 0, stream>>>(wv, wvb, (int)WE);
  cvt_f32_bf16<<<512,  256, 0, stream>>>(wo, wob, (int)WE);

  dim3 ggrid(D_MODEL / 256, M_ROWS / 128);
  gemm_bf16_nt<<<ggrid, 256, 0, stream>>>(xb, wqb, Qb, 0);
  gemm_bf16_nt<<<ggrid, 256, 0, stream>>>(xb, wkb, Kb, 0);
  gemm_bf16_nt<<<ggrid, 256, 0, stream>>>(xb, wvb, Vb, 0);

  attn_flash<<<dim3(BATCH * NUM_HEADS, SEQ / 64), 128, 0, stream>>>(Qb, Kb, Vb, Ob);

  gemm_bf16_nt<<<ggrid, 256, 0, stream>>>(Ob, wob, d_out, 1);
}
